// SlotAttention_37778532336200
// MI455X (gfx1250) — matrix operand order repaired and checked by reading, not compiled, not run on hardware
//
#include <hip/hip_runtime.h>
#include <hip/hip_bf16.h>

// Problem constants
#define BB      64
#define NTOK    4096
#define DIN     384
#define DS      128
#define NSLOT   8
#define HID     256
#define ITERS   3

typedef __attribute__((ext_vector_type(16))) __bf16         v16bf;
typedef __attribute__((ext_vector_type(16))) unsigned short v16us;
typedef __attribute__((ext_vector_type(8)))  unsigned short v8us;
typedef __attribute__((ext_vector_type(8)))  float          v8f;

__device__ __forceinline__ unsigned short f2bf(float x) {
  unsigned u = __float_as_uint(x);
  u += 0x7fffu + ((u >> 16) & 1u);       // round-to-nearest-even
  return (unsigned short)(u >> 16);
}
__device__ __forceinline__ float bf2f(unsigned short u) {
  return __uint_as_float(((unsigned)u) << 16);
}

// CDNA5 async global->LDS copy (ASYNCcnt path), GV addressing mode.
__device__ __forceinline__ void async_g2l_b128(unsigned lds_off, const void* gptr) {
  asm volatile("global_load_async_to_lds_b128 %0, %1, off"
               :: "v"(lds_off), "v"(gptr)
               : "memory");
}
__device__ __forceinline__ void wait_async0() {
  asm volatile("s_wait_asynccnt 0" ::: "memory");
}
__device__ __forceinline__ unsigned lds_off_of(const void* p) {
  // generic LDS address: low 32 bits are the wave-relative LDS byte offset
  return (unsigned)(uintptr_t)p;
}

// ---------------------------------------------------------------------------
// Kernel 0: transpose Wk|Wv -> bf16 WkvT[256][384]  (col-major weights so that
// WMMA B-fragments are contiguous 16-element K runs)
// ---------------------------------------------------------------------------
__global__ void prep_weights(const float* __restrict__ Wk,
                             const float* __restrict__ Wv,
                             unsigned short* __restrict__ wkvT) {
  int idx = blockIdx.x * 256 + threadIdx.x;
  if (idx >= 256 * DIN) return;
  int c = idx / DIN, k = idx % DIN;
  float v = (c < DS) ? Wk[k * DS + c] : Wv[k * DS + (c - DS)];
  wkvT[c * DIN + k] = f2bf(v);
}

// ---------------------------------------------------------------------------
// Kernel 1: fused LayerNorm + [k|v] = LN(f) @ [Wk|Wv]  via WMMA bf16
// One block = 64 feature rows x 256 output cols.  8 waves: wave w owns
// M-tile (w>>1) and 8 N-tiles starting at (w&1)*8.  B panel (256x32) staged
// in LDS per K-step via async loads, double-buffered; A fragment reused
// across the 8 N-tiles (64 accumulator VGPRs).
// ---------------------------------------------------------------------------
#define BPITCH 40  // ushorts per staged B column (32 data + 8 pad; 80B, 16B-mult)

__global__ __launch_bounds__(256) void ln_kv_kernel(
    const float* __restrict__ feat,
    const float* __restrict__ lnw, const float* __restrict__ lnb,
    const unsigned short* __restrict__ wkvT,
    unsigned short* __restrict__ kv) {
  __shared__ __align__(16) float          fs[64 * DIN];      // 96 KB raw rows
  __shared__ __align__(16) unsigned short fb[64 * 392];      // 49 KB LN rows (bf16)
  __shared__ __align__(16) unsigned short Bst[2][256 * BPITCH]; // 2x20 KB B panels
  __shared__ float red[2][64][4];
  __shared__ float mu_s[64], ri_s[64];

  const int    tid     = threadIdx.x;
  const size_t rowbase = (size_t)blockIdx.x * 64;

  // async-load 64 rows x 384 f32 straight into LDS (24 x b128 per thread)
  {
    const unsigned fsb  = lds_off_of(fs);
    const float*   gsrc = feat + rowbase * DIN;
#pragma unroll
    for (int j = 0; j < 24; j++) {
      int lin = tid + j * 256;
      async_g2l_b128(fsb + lin * 16, gsrc + lin * 4);
    }
  }
  // async-stage B panel for K-step 0 (each thread owns one output column)
  {
    const unsigned bb = lds_off_of(&Bst[0][tid * BPITCH]);
    const unsigned short* g = wkvT + tid * DIN;
#pragma unroll
    for (int i = 0; i < 4; i++) async_g2l_b128(bb + i * 16, g + i * 8);
  }
  wait_async0();
  __syncthreads();

  // per-row mean/var: 4 threads per row, 96 elems each
  {
    int   r = tid >> 2, t = tid & 3;
    float s = 0.f, s2 = 0.f;
#pragma unroll
    for (int i = 0; i < 96; i++) {
      float x = fs[r * DIN + t * 96 + i];
      s += x; s2 += x * x;
    }
    red[0][r][t] = s; red[1][r][t] = s2;
  }
  __syncthreads();
  if (tid < 64) {
    float s  = red[0][tid][0] + red[0][tid][1] + red[0][tid][2] + red[0][tid][3];
    float s2 = red[1][tid][0] + red[1][tid][1] + red[1][tid][2] + red[1][tid][3];
    float mu  = s * (1.0f / DIN);
    float var = s2 * (1.0f / DIN) - mu * mu;
    mu_s[tid] = mu;
    ri_s[tid] = rsqrtf(var + 1e-5f);
  }
  __syncthreads();

  // normalize + affine -> bf16 staging (96 elems per thread)
#pragma unroll
  for (int j = 0; j < 96; j++) {
    int   idx = tid + j * 256;
    int   r = idx / DIN, c = idx % DIN;
    float x = (fs[idx] - mu_s[r]) * ri_s[r] * lnw[c] + lnb[c];
    fb[r * 392 + c] = f2bf(x);
  }
  __syncthreads();

  const int lane = tid & 31, wid = tid >> 5;
  const int half = lane >> 4, m = lane & 15;
  const int mt   = wid >> 1;            // M-tile 0..3
  const int ntb  = (wid & 1) * 8;       // first of 8 N-tiles
  const int arow = (mt * 16 + m) * 392;
  const int offA = half ? 16 : 0;

  v8f acc[8];
#pragma unroll
  for (int j = 0; j < 8; j++) acc[j] = (v8f){0.f,0.f,0.f,0.f,0.f,0.f,0.f,0.f};

  int buf = 0;
  for (int kk = 0; kk < 12; kk++) {
    // prefetch next B panel asynchronously while computing this one
    if (kk < 11) {
      const unsigned bb = lds_off_of(&Bst[buf ^ 1][tid * BPITCH]);
      const unsigned short* g = wkvT + tid * DIN + (kk + 1) * 32;
#pragma unroll
      for (int i = 0; i < 4; i++) async_g2l_b128(bb + i * 16, g + i * 8);
    }
    const int k0 = kk * 32;
    // A-fragment (16x32 bf16): lane<16 -> K {0..15}; lane>=16 -> {16..31}
    const v8us lo = *(const v8us*)&fb[arow + k0 + offA];
    const v8us hi = *(const v8us*)&fb[arow + k0 + offA + 8];
    v16us af;
#pragma unroll
    for (int i = 0; i < 8; i++) { af[i] = lo[i]; af[i + 8] = hi[i]; }
    const v16bf A = __builtin_bit_cast(v16bf, af);
#pragma unroll
    for (int j = 0; j < 8; j++) {
      const int   col = (ntb + j) * 16 + m;
      const v16us bfv = *(const v16us*)&Bst[buf][col * BPITCH + (half ? 16 : 0)];
      acc[j] = __builtin_amdgcn_wmma_f32_16x16x32_bf16(
          false, A, false, __builtin_bit_cast(v16bf, bfv),
          (short)0, acc[j], false, false);
    }
    wait_async0();
    __syncthreads();
    buf ^= 1;
  }

  // C layout: VGPR jj -> M = jj + (half?8:0), N = lane&15
#pragma unroll
  for (int j = 0; j < 8; j++) {
    const int col = (ntb + j) * 16 + m;
#pragma unroll
    for (int jj = 0; jj < 8; jj++) {
      const size_t row = rowbase + mt * 16 + jj + (half ? 8 : 0);
      kv[row * 256 + col] = f2bf(acc[j][jj]);
    }
  }
}

// ---------------------------------------------------------------------------
// Kernel 2: per-batch slot-attention iterations (persistent block per batch)
// ---------------------------------------------------------------------------
__global__ __launch_bounds__(256) void slot_attn_kernel(
    const unsigned short* __restrict__ kv,
    const float* __restrict__ Wq,
    const float* __restrict__ lsw, const float* __restrict__ lsb,
    const float* __restrict__ Wih, const float* __restrict__ Whh,
    const float* __restrict__ bih, const float* __restrict__ bhh,
    const float* __restrict__ mlw, const float* __restrict__ mlb,
    const float* __restrict__ W1, const float* __restrict__ b1,
    const float* __restrict__ W2, const float* __restrict__ b2,
    const float* __restrict__ slot_mu,
    float* __restrict__ out) {
  __shared__ float                         attn_l[NSLOT * NTOK];  // 128 KB
  __shared__ __align__(16) unsigned short  kst[256 * 128];        // 64 KB, swizzled
  __shared__ float qld[NSLOT * DS];
  __shared__ float sbuf[NSLOT * DS];
  __shared__ float ubuf[NSLOT * DS];
  __shared__ float mbuf[NSLOT * DS];
  __shared__ float gi[NSLOT * 3 * DS];
  __shared__ float gh[NSLOT * 3 * DS];
  __shared__ float hid[NSLOT * HID];
  __shared__ float rowsum[NSLOT];
  __shared__ float rinv[NSLOT];

  const int    tid   = threadIdx.x;
  const int    lane  = tid & 31;
  const int    b     = blockIdx.x;
  const size_t kvb   = (size_t)b * NTOK * 256;
  const float  scale = 0.08838834764831845f;  // 128^-0.5

#pragma unroll
  for (int j = 0; j < 4; j++) { int o = tid + j * 256; sbuf[o] = slot_mu[o & 127]; }
  __syncthreads();

  for (int it = 0; it < ITERS; it++) {
    // ---- s = LN(slots) -> mbuf  (one wave32 per slot row)
    {
      int   r = tid >> 5;
      float s = 0.f, s2 = 0.f;
#pragma unroll
      for (int i = 0; i < 4; i++) {
        float x = sbuf[r * DS + lane * 4 + i];
        s += x; s2 += x * x;
      }
#pragma unroll
      for (int off = 16; off > 0; off >>= 1) {
        s  += __shfl_xor(s, off, 32);
        s2 += __shfl_xor(s2, off, 32);
      }
      float mu = s * (1.f / DS);
      float ri = rsqrtf(s2 * (1.f / DS) - mu * mu + 1e-5f);
#pragma unroll
      for (int i = 0; i < 4; i++) {
        int c = lane * 4 + i;
        mbuf[r * DS + c] = (sbuf[r * DS + c] - mu) * ri * lsw[c] + lsb[c];
      }
    }
    __syncthreads();
    // ---- q = scale * s @ Wq
#pragma unroll
    for (int j = 0; j < 4; j++) {
      int   o = tid + j * 256, s = o >> 7, d = o & 127;
      float a = 0.f;
      for (int k = 0; k < DS; k++) a += mbuf[s * DS + k] * Wq[k * DS + d];
      qld[o] = a * scale;
    }
    if (tid < NSLOT) rowsum[tid] = 0.f;
    __syncthreads();

    // ---- pass 1: logits, softmax over slots, attn -> LDS, rowsum
    for (int ch = 0; ch < NTOK / 256; ch++) {
      const int n0 = ch * 256;
      // async-stage k-half of 256 tokens; XOR-swizzle 16B sub-chunks so the
      // lane-per-row dword reads below avoid bank conflicts
      {
#pragma unroll
        for (int j = 0; j < 16; j++) {
          int lin = tid + j * 256;
          int r = lin >> 4, c = lin & 15;
          unsigned loff = lds_off_of(&kst[r * 128 + ((c ^ (r & 15)) * 8)]);
          async_g2l_b128(loff, kv + kvb + (size_t)(n0 + r) * 256 + c * 8);
        }
      }
      wait_async0();
      __syncthreads();

      const int             n    = n0 + tid;
      const unsigned short* krow = kst + tid * 128;
      float lg[NSLOT];
#pragma unroll
      for (int s = 0; s < NSLOT; s++) lg[s] = 0.f;
#pragma unroll 4
      for (int cc = 0; cc < 16; cc++) {
        const unsigned* cp = (const unsigned*)(krow + ((cc ^ (tid & 15)) * 8));
#pragma unroll
        for (int w = 0; w < 4; w++) {
          unsigned pk = cp[w];
          int      d0 = cc * 8 + w * 2;
          float    k0 = bf2f((unsigned short)pk);
          float    k1 = bf2f((unsigned short)(pk >> 16));
#pragma unroll
          for (int s = 0; s < NSLOT; s++)
            lg[s] += qld[s * DS + d0] * k0 + qld[s * DS + d0 + 1] * k1;
        }
      }
      float mx = lg[0];
#pragma unroll
      for (int s = 1; s < NSLOT; s++) mx = fmaxf(mx, lg[s]);
      float e[NSLOT], tot = 0.f;
#pragma unroll
      for (int s = 0; s < NSLOT; s++) { e[s] = __expf(lg[s] - mx); tot += e[s]; }
      float itot = 1.f / tot;
#pragma unroll
      for (int s = 0; s < NSLOT; s++) {
        float a = e[s] * itot;
        attn_l[s * NTOK + n] = a;
        float v = a;
#pragma unroll
        for (int off = 16; off > 0; off >>= 1) v += __shfl_xor(v, off, 32);
        if (lane == 0) atomicAdd(&rowsum[s], v);
      }
      __syncthreads();
    }
    if (tid < NSLOT) rinv[tid] = 1.f / (rowsum[tid] + 1e-8f);
    __syncthreads();

    // last iteration: emit raw softmax attn (reference returns pre-norm attn)
    if (it == ITERS - 1) {
      float* oa = out + BB * NSLOT * DS + (size_t)b * NSLOT * NTOK;
#pragma unroll 4
      for (int j = 0; j < 128; j++) { int o = tid + j * 256; oa[o] = attn_l[o]; }
    }

    // ---- updates = (attn / rowsum) @ v   (1/rowsum factored outside sum)
    {
      const int             d  = tid & 127, sg = tid >> 7;
      const unsigned short* vp = kv + kvb + 128 + d;
      const float*          ap = attn_l + (sg * 4) * NTOK;
      float u0 = 0.f, u1 = 0.f, u2 = 0.f, u3 = 0.f;
#pragma unroll 4
      for (int n = 0; n < NTOK; n++) {
        float vv = bf2f(vp[(size_t)n * 256]);
        u0 += ap[n] * vv;
        u1 += ap[NTOK + n] * vv;
        u2 += ap[2 * NTOK + n] * vv;
        u3 += ap[3 * NTOK + n] * vv;
      }
      ubuf[(sg * 4 + 0) * DS + d] = u0 * rinv[sg * 4 + 0];
      ubuf[(sg * 4 + 1) * DS + d] = u1 * rinv[sg * 4 + 1];
      ubuf[(sg * 4 + 2) * DS + d] = u2 * rinv[sg * 4 + 2];
      ubuf[(sg * 4 + 3) * DS + d] = u3 * rinv[sg * 4 + 3];
    }
    __syncthreads();

    // ---- GRU cell gates: gi = upd@Wih + bih, gh = slots@Whh + bhh
#pragma unroll
    for (int j = 0; j < 12; j++) {
      int   o = tid + j * 256, s = o / 384, c = o % 384;
      float a = bih[c], h2 = bhh[c];
      for (int k = 0; k < DS; k++) {
        a  += ubuf[s * DS + k] * Wih[k * 384 + c];
        h2 += sbuf[s * DS + k] * Whh[k * 384 + c];
      }
      gi[o] = a; gh[o] = h2;
    }
    __syncthreads();
#pragma unroll
    for (int j = 0; j < 4; j++) {
      int   o = tid + j * 256, s = o >> 7, d = o & 127;
      float r  = 1.f / (1.f + __expf(-(gi[s * 384 + d] + gh[s * 384 + d])));
      float z  = 1.f / (1.f + __expf(-(gi[s * 384 + 128 + d] + gh[s * 384 + 128 + d])));
      float nn = tanhf(gi[s * 384 + 256 + d] + r * gh[s * 384 + 256 + d]);
      sbuf[o] = (1.f - z) * nn + z * sbuf[o];
    }
    __syncthreads();

    // ---- residual MLP: slots += silu(LN(slots)@W1+b1)@W2+b2
    {
      int   r = tid >> 5;
      float s = 0.f, s2 = 0.f;
#pragma unroll
      for (int i = 0; i < 4; i++) {
        float x = sbuf[r * DS + lane * 4 + i];
        s += x; s2 += x * x;
      }
#pragma unroll
      for (int off = 16; off > 0; off >>= 1) {
        s  += __shfl_xor(s, off, 32);
        s2 += __shfl_xor(s2, off, 32);
      }
      float mu = s * (1.f / DS);
      float ri = rsqrtf(s2 * (1.f / DS) - mu * mu + 1e-5f);
#pragma unroll
      for (int i = 0; i < 4; i++) {
        int c = lane * 4 + i;
        mbuf[r * DS + c] = (sbuf[r * DS + c] - mu) * ri * mlw[c] + mlb[c];
      }
    }
    __syncthreads();
#pragma unroll
    for (int j = 0; j < 8; j++) {
      int   o = tid + j * 256, s = o >> 8, c = o & 255;
      float a = b1[c];
      for (int k = 0; k < DS; k++) a += mbuf[s * DS + k] * W1[k * HID + c];
      hid[o] = a * (1.f / (1.f + __expf(-a)));  // silu
    }
    __syncthreads();
#pragma unroll
    for (int j = 0; j < 4; j++) {
      int   o = tid + j * 256, s = o >> 7, d = o & 127;
      float a = b2[d];
      for (int k = 0; k < HID; k++) a += hid[s * HID + k] * W2[k * DS + d];
      sbuf[o] += a;
    }
    __syncthreads();
  }

  // final slots
#pragma unroll
  for (int j = 0; j < 4; j++) {
    int o = tid + j * 256;
    out[(size_t)b * NSLOT * DS + o] = sbuf[o];
  }
}

// ---------------------------------------------------------------------------
extern "C" void kernel_launch(void* const* d_in, const int* in_sizes, int n_in,
                              void* d_out, int out_size, void* d_ws, size_t ws_size,
                              hipStream_t stream) {
  const float* features = (const float*)d_in[0];
  const float* ln_in_w  = (const float*)d_in[1];
  const float* ln_in_b  = (const float*)d_in[2];
  const float* Wk       = (const float*)d_in[3];
  const float* Wv       = (const float*)d_in[4];
  const float* Wq       = (const float*)d_in[5];
  const float* ln_s_w   = (const float*)d_in[6];
  const float* ln_s_b   = (const float*)d_in[7];
  const float* gru_Wih  = (const float*)d_in[8];
  const float* gru_Whh  = (const float*)d_in[9];
  const float* gru_bih  = (const float*)d_in[10];
  const float* gru_bhh  = (const float*)d_in[11];
  const float* mlp_ln_w = (const float*)d_in[12];
  const float* mlp_ln_b = (const float*)d_in[13];
  const float* mlp_W1   = (const float*)d_in[14];
  const float* mlp_b1   = (const float*)d_in[15];
  const float* mlp_W2   = (const float*)d_in[16];
  const float* mlp_b2   = (const float*)d_in[17];
  const float* slot_mu  = (const float*)d_in[18];

  // workspace: WkvT bf16 [256][384] (192 KB) then kv bf16 [B*N][256] (128 MB)
  unsigned short* wkvT = (unsigned short*)d_ws;
  unsigned short* kv   = wkvT + 256 * DIN;

  prep_weights<<<(256 * DIN + 255) / 256, 256, 0, stream>>>(Wk, Wv, wkvT);
  ln_kv_kernel<<<(BB * NTOK) / 64, 256, 0, stream>>>(features, ln_in_w, ln_in_b,
                                                     wkvT, kv);
  slot_attn_kernel<<<BB, 256, 0, stream>>>(
      kv, Wq, ln_s_w, ln_s_b, gru_Wih, gru_Whh, gru_bih, gru_bhh,
      mlp_ln_w, mlp_ln_b, mlp_W1, mlp_b1, mlp_W2, mlp_b2, slot_mu,
      (float*)d_out);
}
